// NonLocalAttention_20925080666260
// MI455X (gfx1250) — matrix operand order repaired and checked by reading, not compiled, not run on hardware
//
#include <hip/hip_runtime.h>

// ---------------------------------------------------------------------------
// NonLocalAttention for MI455X (gfx1250, wave32, WMMA).
// Compute-bound (~275 GFLOP vs ~8us of HBM traffic at 23.3 TB/s), so all
// GEMMs run on v_wmma_f32_16x16x32_bf16 (bf16 in, f32 accumulate); softmax
// logits and the final output stay f32.
// GEMM core: block = 8 waves = 128(M) x 64(N) tile; per-wave 16x64 -> four
// 16x16 f32 accumulators. B panel (64x32 bf16) ping-pong buffered in LDS with
// padded rows (40 bf16 = 20 banks), staged divergence-free (1 uint4/thread).
// The ping-pong is unrolled explicitly (64 K per iteration) so both LDS bases
// are loop-invariant and each phase is an unconditional stage + 4-WMMA burst.
// Workspace (~152 MB):
//   [0  , 32M) XT  bf16 [N][HW][C]   (dead after convs; S/P reuse it)
//   [32M, 56M) W1/2/3 bf16 [C][C]
//   [56M, 88M) Q   bf16 [N][HW][C]
//   [88M,120M) KT  bf16 [N][HW][C]
//   [120M,152M) V  bf16 [N][C][HW]
//   S f32 [HW][HW] (4MB) @0;  P bf16 [HW][HW] (2MB) @4MB
// ---------------------------------------------------------------------------

#define NB 8
#define C  2048
#define HW 1024

typedef __bf16 bf16_t;
typedef __attribute__((ext_vector_type(8)))  __bf16 v8bf;
typedef __attribute__((ext_vector_type(16))) __bf16 v16bf;
typedef __attribute__((ext_vector_type(4)))  float  v4f;
typedef __attribute__((ext_vector_type(8)))  float  v8f;

__device__ __forceinline__ v16bf cat16(v8bf lo, v8bf hi) {
  return __builtin_shufflevector(lo, hi, 0,1,2,3,4,5,6,7,8,9,10,11,12,13,14,15);
}

__device__ __forceinline__ float celu1(float y) {
  return y > 0.0f ? y : (__expf(y) - 1.0f);  // CELU, alpha=1
}

// One k-step: 1 A fragment (global), 4 B fragments (LDS, distinct regs),
// then a 4-WMMA burst. Fragment layouts per ISA 7.12.2:
//   A: row = lane&15, K chunks at ((lane>>4)<<3) and +16
//   B: col = lane&15, K chunks at ((lane>>4)<<3) and +16
__device__ __forceinline__ void wmma_step(
    const bf16_t* __restrict__ arow, const bf16_t* __restrict__ lds,
    const int k, const int kA, const int kB, const int l15, v8f (&acc)[4]) {
  v16bf a = cat16(*(const v8bf*)(arow + k + kA),
                  *(const v8bf*)(arow + k + kA + 16));
  v16bf b[4];
#pragma unroll
  for (int j = 0; j < 4; ++j) {
    const bf16_t* bp = lds + (j * 16 + l15) * 40 + kB;
    b[j] = cat16(*(const v8bf*)bp, *(const v8bf*)(bp + 16));
  }
#pragma unroll
  for (int j = 0; j < 4; ++j)
    acc[j] = __builtin_amdgcn_wmma_f32_16x16x32_bf16(false, a, false, b[j],
                                                     (short)0, acc[j], false, false);
}

// ---------------------------------------------------------------------------
// Shared GEMM core. Wave computes D[16 x 64]:
//   arow   : &A[(m0 + lane&15) * K]   (16 rows, contiguous K)   [global]
//   bstage : &B[n0 * K]               (64 rows, contiguous K)   [global]
// K must be a multiple of 64 (here: 2048 or 1024).
// ---------------------------------------------------------------------------
__device__ __forceinline__ void gemm4_core(
    const bf16_t* __restrict__ arow, const bf16_t* __restrict__ bstage,
    const int K, const int lane, const int tid,
    uint4* __restrict__ lds4, v8f (&acc)[4]) {
  const int kA  = (lane >> 4) << 3;
  const int kB  = (lane >> 4) << 3;
  const int l15 = lane & 15;
  // Each of the 256 threads stages one uint4 (8 bf16): row tid>>2, chunk tid&3.
  const bf16_t* sptr = bstage + (size_t)(tid >> 2) * K + (tid & 3) * 8;
  const int si = (tid >> 2) * 5 + (tid & 3);  // padded row = 5 uint4 (4 + 1 pad)
  uint4* buf0 = lds4;
  uint4* buf1 = lds4 + 320;
  const bf16_t* l0 = (const bf16_t*)buf0;
  const bf16_t* l1 = (const bf16_t*)buf1;

  buf0[si] = *(const uint4*)sptr;
  __syncthreads();

  uint4 st;
  int k = 0;
  for (; k < K - 64; k += 64) {
    // phase 0: compute from buf0, stage panel k+32 into buf1
    st = *(const uint4*)(sptr + k + 32);
    __builtin_prefetch(arow + k + 192, 0, 1);  // speculative A-stream prefetch
    wmma_step(arow, l0, k, kA, kB, l15, acc);
    buf1[si] = st;
    __syncthreads();
    // phase 1: compute from buf1, stage panel k+64 into buf0
    st = *(const uint4*)(sptr + k + 64);
    wmma_step(arow, l1, k + 32, kA, kB, l15, acc);
    buf0[si] = st;
    __syncthreads();
  }
  // tail (k == K-64): stage last panel, two final bursts, no trailing barrier
  st = *(const uint4*)(sptr + k + 32);
  wmma_step(arow, l0, k, kA, kB, l15, acc);
  buf1[si] = st;
  __syncthreads();
  wmma_step(arow, l1, k + 32, kA, kB, l15, acc);
}

// ---- x [N,C,HW] f32 -> XT [N,HW,C] bf16 (LDS tile transpose) ---------------
__global__ __launch_bounds__(256) void k_transpose_cvt(
    const float* __restrict__ x, bf16_t* __restrict__ xt) {
  __shared__ float tile[32][33];
  const int n  = blockIdx.z;
  const int c0 = blockIdx.y * 32;
  const int t0 = blockIdx.x * 32;
  const int tx = threadIdx.x, ty = threadIdx.y;  // 32 x 8
  const float* src = x + ((size_t)n * C + c0) * HW + t0;
#pragma unroll
  for (int i = 0; i < 32; i += 8) tile[ty + i][tx] = src[(size_t)(ty + i) * HW + tx];
  __syncthreads();
  bf16_t* dst = xt + ((size_t)n * HW + t0) * C + c0;
#pragma unroll
  for (int i = 0; i < 32; i += 8) dst[(size_t)(ty + i) * C + tx] = (bf16_t)tile[tx][ty + i];
}

// ---- f32 -> bf16 elementwise (weights) -------------------------------------
__global__ __launch_bounds__(256) void k_cvt_bf16(
    const float* __restrict__ in, bf16_t* __restrict__ out, int nelem) {
  int idx = blockIdx.x * 256 + threadIdx.x;
  const int stride = gridDim.x * 256;
  for (; idx < nelem; idx += stride) out[idx] = (bf16_t)in[idx];
}

// ---- 1x1 conv GEMM: E[o,t] = celu( sum_c W[o,c]*XT[t,c] + b[o] ) -----------
// TRANS_OUT: store [N,HW,C] (Q, K^T); else [N,C,HW] (V).
template <bool TRANS_OUT>
__global__ __launch_bounds__(256) void k_conv_gemm(
    const bf16_t* __restrict__ xt, const bf16_t* __restrict__ w,
    const float* __restrict__ bias, bf16_t* __restrict__ out) {
  __shared__ uint4 lds4[640];
  const int tid = threadIdx.x, lane = tid & 31, wave = tid >> 5;
  const int n  = blockIdx.z;
  const int t0 = blockIdx.x * 64;                 // N (tokens)
  const int o0 = blockIdx.y * 128 + wave * 16;    // M (out channels)
  const bf16_t* bstage = xt + ((size_t)n * HW + t0) * C;
  const bf16_t* arow   = w + (size_t)(o0 + (lane & 15)) * C;

  v8f acc[4] = {};
  gemm4_core(arow, bstage, C, lane, tid, lds4, acc);

  // D tile: lane holds M = i + 8*(lane>>4) (o), N = lane&15 (t)
  const int obase = o0 + ((lane >> 4) << 3);
  v4f b0 = *(const v4f*)(bias + obase);
  v4f b1 = *(const v4f*)(bias + obase + 4);
#pragma unroll
  for (int j = 0; j < 4; ++j) {
    const int tcol = t0 + j * 16 + (lane & 15);
    if (TRANS_OUT) {
      v8bf r;
#pragma unroll
      for (int i = 0; i < 8; ++i) {
        float y = acc[j][i] + (i < 4 ? b0[i] : b1[i - 4]);
        r[i] = (bf16_t)celu1(y);
      }
      *(v8bf*)(out + ((size_t)n * HW + tcol) * C + obase) = r;  // 16B store
    } else {
#pragma unroll
      for (int i = 0; i < 8; ++i) {
        float y = acc[j][i] + (i < 4 ? b0[i] : b1[i - 4]);
        out[((size_t)n * C + obase + i) * HW + tcol] = (bf16_t)celu1(y);
      }
    }
  }
}

// ---- S[t,s] = sum_c Q[t,c] * KT[s,c] (f32 logits) --------------------------
__global__ __launch_bounds__(256) void k_score_gemm(
    const bf16_t* __restrict__ q, const bf16_t* __restrict__ kt,
    float* __restrict__ s) {
  __shared__ uint4 lds4[640];
  const int tid = threadIdx.x, lane = tid & 31, wave = tid >> 5;
  const int s0 = blockIdx.x * 64;                 // N (keys)
  const int t0 = blockIdx.y * 128 + wave * 16;    // M (queries)
  const bf16_t* arow   = q + (size_t)(t0 + (lane & 15)) * C;
  const bf16_t* bstage = kt + (size_t)s0 * C;

  v8f acc[4] = {};
  gemm4_core(arow, bstage, C, lane, tid, lds4, acc);

  const int tbase = t0 + ((lane >> 4) << 3);
#pragma unroll
  for (int j = 0; j < 4; ++j) {
    const int scol = s0 + j * 16 + (lane & 15);
#pragma unroll
    for (int i = 0; i < 8; ++i) s[(size_t)(tbase + i) * HW + scol] = acc[j][i];
  }
}

// ---- row softmax over 1024, f32 logits -> bf16 probabilities ---------------
__global__ __launch_bounds__(256) void k_softmax(
    const float* __restrict__ S, bf16_t* __restrict__ P) {
  const int t   = blockIdx.x;
  const int tid = threadIdx.x;
  const float* row = S + (size_t)t * HW;
  __shared__ float red[256];
  float v[4];
  float m = -3.402823466e38f;
#pragma unroll
  for (int i = 0; i < 4; ++i) { v[i] = row[tid + 256 * i]; m = fmaxf(m, v[i]); }
  red[tid] = m;
  __syncthreads();
  for (int off = 128; off > 0; off >>= 1) {
    if (tid < off) red[tid] = fmaxf(red[tid], red[tid + off]);
    __syncthreads();
  }
  m = red[0];
  __syncthreads();
  float sum = 0.0f;
#pragma unroll
  for (int i = 0; i < 4; ++i) { v[i] = __expf(v[i] - m); sum += v[i]; }
  red[tid] = sum;
  __syncthreads();
  for (int off = 128; off > 0; off >>= 1) {
    if (tid < off) red[tid] += red[tid + off];
    __syncthreads();
  }
  const float inv = 1.0f / red[0];
  bf16_t* prow = P + (size_t)t * HW;
#pragma unroll
  for (int i = 0; i < 4; ++i) prow[tid + 256 * i] = (bf16_t)(v[i] * inv);
}

// ---- out[c,t] = sum_s P[t,s] * V[c,s]; f32 straight into [n,c,h,w] ---------
__global__ __launch_bounds__(256) void k_out_gemm(
    const bf16_t* __restrict__ p, const bf16_t* __restrict__ v,
    float* __restrict__ out) {
  __shared__ uint4 lds4[640];
  const int tid = threadIdx.x, lane = tid & 31, wave = tid >> 5;
  const int c0 = blockIdx.x * 64;                 // N (channels)
  const int t0 = blockIdx.y * 128 + wave * 16;    // M (tokens)
  const bf16_t* arow   = p + (size_t)(t0 + (lane & 15)) * HW;
  const bf16_t* bstage = v + (size_t)c0 * HW;

  v8f acc[4] = {};
  gemm4_core(arow, bstage, HW, lane, tid, lds4, acc);

  // D: M = t (8 contiguous per lane), N = c -> coalesced v4f stores
  const int tb = t0 + ((lane >> 4) << 3);
#pragma unroll
  for (int j = 0; j < 4; ++j) {
    const int c = c0 + j * 16 + (lane & 15);
    float* dst = out + (size_t)c * HW + tb;
    v4f lo = {acc[j][0], acc[j][1], acc[j][2], acc[j][3]};
    v4f hi = {acc[j][4], acc[j][5], acc[j][6], acc[j][7]};
    *(v4f*)dst = lo;
    *(v4f*)(dst + 4) = hi;
  }
}

extern "C" void kernel_launch(void* const* d_in, const int* in_sizes, int n_in,
                              void* d_out, int out_size, void* d_ws, size_t ws_size,
                              hipStream_t stream) {
  const float* x  = (const float*)d_in[0];
  const float* w1 = (const float*)d_in[1];
  const float* b1 = (const float*)d_in[2];
  const float* w2 = (const float*)d_in[3];
  const float* b2 = (const float*)d_in[4];
  const float* w3 = (const float*)d_in[5];
  const float* b3 = (const float*)d_in[6];
  float* out = (float*)d_out;
  char* ws = (char*)d_ws;

  const size_t MB = 1024ull * 1024ull;
  bf16_t* XT  = (bf16_t*)(ws + 0);          // 32 MB
  bf16_t* W1b = (bf16_t*)(ws + 32 * MB);    // 8 MB each
  bf16_t* W2b = (bf16_t*)(ws + 40 * MB);
  bf16_t* W3b = (bf16_t*)(ws + 48 * MB);
  bf16_t* Q   = (bf16_t*)(ws + 56 * MB);    // 32 MB
  bf16_t* KT  = (bf16_t*)(ws + 88 * MB);    // 32 MB
  bf16_t* V   = (bf16_t*)(ws + 120 * MB);   // 32 MB
  // XT region is dead after the conv GEMMs: reuse it for per-batch S and P.
  float*  S = (float*)(ws + 0);             // 4 MB
  bf16_t* P = (bf16_t*)(ws + 4 * MB);       // 2 MB

  // 1) pack inputs to bf16
  k_transpose_cvt<<<dim3(HW / 32, C / 32, NB), dim3(32, 8), 0, stream>>>(x, XT);
  k_cvt_bf16<<<4096, 256, 0, stream>>>(w1, W1b, C * C);
  k_cvt_bf16<<<4096, 256, 0, stream>>>(w2, W2b, C * C);
  k_cvt_bf16<<<4096, 256, 0, stream>>>(w3, W3b, C * C);

  // 2) three 1x1-conv GEMMs (bias + CELU fused)
  dim3 gConv(HW / 64, C / 128, NB);
  k_conv_gemm<true ><<<gConv, 256, 0, stream>>>(XT, W1b, b1, Q);   // q  [n,t,c]
  k_conv_gemm<true ><<<gConv, 256, 0, stream>>>(XT, W2b, b2, KT);  // k^T[n,s,c]
  k_conv_gemm<false><<<gConv, 256, 0, stream>>>(XT, W3b, b3, V);   // v^T[n,c,s]

  // 3) per-batch attention: S = QK^T, P = softmax(S), out = (PV)^T
  for (int n = 0; n < NB; ++n) {
    const bf16_t* Qn = Q  + (size_t)n * HW * C;
    const bf16_t* Kn = KT + (size_t)n * HW * C;
    const bf16_t* Vn = V  + (size_t)n * C * HW;
    float* On = out + (size_t)n * C * HW;
    k_score_gemm<<<dim3(HW / 64, HW / 128), 256, 0, stream>>>(Qn, Kn, S);
    k_softmax  <<<HW, 256, 0, stream>>>(S, P);
    k_out_gemm <<<dim3(C / 64, HW / 128), 256, 0, stream>>>(P, Vn, On);
  }
}
